// CGCNNConv_4690104287279
// MI455X (gfx1250) — matrix operand order repaired and checked by reading, not compiled, not run on hardware
//
#include <hip/hip_runtime.h>
#include <math.h>

// ---------------------------------------------------------------------------
// CGCNN conv fused kernel for gfx1250 (MI455X).
// GEMM path: split-bf16 (hi+lo) V_WMMA_F32_16X16X32_BF16, 3 WMMAs per tile
// product for ~fp32 accuracy at bf16 matrix-pipe rate.
// fc_w is pre-split once into bf16 hi/lo planes in d_ws (removes per-block
// redundant conversion VALU from the hot loop); falls back to in-loop split
// if the workspace is too small.
// ---------------------------------------------------------------------------

typedef __attribute__((ext_vector_type(16))) __bf16 v16bf;
typedef __attribute__((ext_vector_type(8)))  float  v8f;

#define ATOMS_PER_BLOCK 4
#define MNBR    12
#define ROWS    48          // 4 atoms * 12 neighbor rows (exactly 3 M-tiles)
#define ADIM    256
#define BDIM    128
#define DDIM    640         // 2A + B  (K of the GEMM)
#define KOUT    512         // 2A      (N of the GEMM)
#define KC      32          // K per bf16 WMMA
#define NKC     20          // 640 / 32
#define THREADS 512
#define NWAVES  16
#define MT      3           // M-tiles per wave (all of them)
#define NT      2           // N-tiles per wave (16 waves * 32 cols = 512)

#define ASTRIDE 648         // bf16 elems per LDS row: 640 + 8 pad (324 dw, %64==4 -> conflict-free)
#define ZSTRIDE 516         // f32 elems per z row
#define OSTRIDE 260         // f32 elems per partial-out slot

#define SMEM_A_BYTES  (2 * ROWS * ASTRIDE * 2)               // hi + lo bf16 planes: 124416 B
#define SMEM_O_OFF    SMEM_A_BYTES
#define SMEM_BYTES    (SMEM_A_BYTES + NWAVES * OSTRIDE * 4)  // + 16 partial slots: 141056 B

#define WPLANE_ELEMS  (KOUT * DDIM)                          // 327680
#define WS_NEEDED     ((size_t)WPLANE_ELEMS * 2 * 2)         // hi+lo ushort planes: 1310720 B

__device__ __forceinline__ unsigned short f32_to_bf16_rne(float f) {
    unsigned int u = __float_as_uint(f);
    unsigned int r = u + 0x7FFFu + ((u >> 16) & 1u);
    return (unsigned short)(r >> 16);
}

// x ~= hi + lo ; hi = truncated bf16 (exact top 16 bits), lo = RNE bf16 of residual
__device__ __forceinline__ void split_bf16(float x, unsigned short& hi, unsigned short& lo) {
    unsigned int u = __float_as_uint(x);
    hi = (unsigned short)(u >> 16);
    float hif = __uint_as_float(u & 0xFFFF0000u);
    lo = f32_to_bf16_rne(x - hif);
}

__device__ __forceinline__ float wave_sum(float v) {
#pragma unroll
    for (int m = 16; m > 0; m >>= 1) v += __shfl_xor(v, m, 32);
    return v;
}

__device__ __forceinline__ float sigmoidf_(float x) { return 1.0f / (1.0f + expf(-x)); }
__device__ __forceinline__ float softplusf_(float x) { return (x > 20.0f) ? x : log1pf(expf(x)); }

union AFrag { uint4 q[2]; v16bf v; };          // two 16B chunks -> one v16bf
union BFrag { unsigned short s[16]; v16bf v; };

// ---- pre-pass: split fc_w into bf16 hi/lo planes (run once per launch) -----
__global__ void split_fcw_kernel(const float* __restrict__ fc_w,
                                 unsigned short* __restrict__ w_hi,
                                 unsigned short* __restrict__ w_lo)
{
    const int i = blockIdx.x * blockDim.x + threadIdx.x;
    if (i < WPLANE_ELEMS) {
        unsigned short h, l;
        split_bf16(fc_w[i], h, l);
        w_hi[i] = h;
        w_lo[i] = l;
    }
}

template <bool PRESPLIT>
__global__ __launch_bounds__(THREADS, 1)
void cgcnn_conv_kernel(const float* __restrict__ atom_feats,
                       const float* __restrict__ bond_feats,
                       const float* __restrict__ fc_w,
                       const unsigned short* __restrict__ w_hi,
                       const unsigned short* __restrict__ w_lo,
                       const float* __restrict__ fc_b,
                       const float* __restrict__ ln1_g, const float* __restrict__ ln1_b,
                       const float* __restrict__ ln2_g, const float* __restrict__ ln2_b,
                       const int*   __restrict__ nbr_idx,
                       float* __restrict__ out, int N)
{
    extern __shared__ __align__(16) char smem[];
    unsigned short* a_hi = (unsigned short*)smem;                       // [ROWS][ASTRIDE]
    unsigned short* a_lo = a_hi + ROWS * ASTRIDE;                       // [ROWS][ASTRIDE]
    float*          zbuf = (float*)smem;                                // reused after GEMM: [ROWS][ZSTRIDE]
    float*          obuf = (float*)(smem + SMEM_O_OFF);                 // [NWAVES][OSTRIDE] partial sums

    const int tid   = threadIdx.x;
    const int lane  = tid & 31;
    const int wid   = tid >> 5;        // 0..15
    const int rsel  = lane & 15;
    const int ksel  = lane >> 4;       // 0 or 1
    const int nbase = blockIdx.x * ATOMS_PER_BLOCK;

    // ---- Phase 1: stage combined [48][640] matrix in LDS as bf16 hi/lo -----
#pragma unroll
    for (int i = 0; i < 3; ++i) {
        const int rr = wid * 3 + i;               // combined row 0..47
        const int a  = rr / MNBR;
        const int m  = rr - a * MNBR;
        const int n  = nbase + a;
        const bool valid = (n < N);
        int nb = 0;
        if (valid) {
            nb = nbr_idx[n * MNBR + m];
            nb = (nb < 0) ? 0 : ((nb >= N) ? N - 1 : nb);
        }
        unsigned short* rh = a_hi + rr * ASTRIDE;
        unsigned short* rl = a_lo + rr * ASTRIDE;
#pragma unroll
        for (int j = 0; j < NKC; ++j) {
            const int d = lane + 32 * j;          // 0..639
            float x = 0.0f;
            if (valid) {
                if (d < ADIM)            x = atom_feats[n * ADIM + d];
                else if (d < 2 * ADIM)   x = atom_feats[nb * ADIM + (d - ADIM)];
                else                     x = bond_feats[(size_t)(n * MNBR + m) * BDIM + (d - 2 * ADIM)];
            }
            unsigned short h, l;
            split_bf16(x, h, l);
            rh[d] = h;
            rl[d] = l;
        }
    }
    __syncthreads();

    // ---- Phase 2: GEMM, D = combined(48x640) x fc_w^T(640x512) -------------
    const int ncol0 = wid * (NT * 16);            // this wave's first output column
    v8f acc[MT][NT] = {};

    for (int kc = 0; kc < NKC; ++kc) {
        // A fragments from LDS (contiguous K: lanes<16 K 0-15, lanes>=16 K 16-31)
        AFrag ah[MT], al[MT];
        const int d0 = kc * KC + ksel * 16;
#pragma unroll
        for (int mt = 0; mt < MT; ++mt) {
            const unsigned short* ph = a_hi + (mt * 16 + rsel) * ASTRIDE;
            const unsigned short* pl = a_lo + (mt * 16 + rsel) * ASTRIDE;
            ah[mt].q[0] = *(const uint4*)(ph + d0);
            ah[mt].q[1] = *(const uint4*)(ph + d0 + 8);
            al[mt].q[0] = *(const uint4*)(pl + d0);
            al[mt].q[1] = *(const uint4*)(pl + d0 + 8);
        }
        // B fragments (B layout: lane holds col rsel, 16 consecutive K)
        AFrag bh[NT], bl[NT];                      // reuse uint4-backed union
        const int koff = kc * KC + ksel * 16;
#pragma unroll
        for (int nt = 0; nt < NT; ++nt) {
            const int col = ncol0 + nt * 16 + rsel;
            if (PRESPLIT) {
                const uint4* ph4 = (const uint4*)(w_hi + (size_t)col * DDIM + koff);
                const uint4* pl4 = (const uint4*)(w_lo + (size_t)col * DDIM + koff);
                bh[nt].q[0] = ph4[0]; bh[nt].q[1] = ph4[1];
                bl[nt].q[0] = pl4[0]; bl[nt].q[1] = pl4[1];
            } else {
                const float4* wp = (const float4*)(fc_w + (size_t)col * DDIM + koff);
                float w[16];
#pragma unroll
                for (int q = 0; q < 4; ++q) {
                    float4 f = wp[q];
                    w[4 * q + 0] = f.x; w[4 * q + 1] = f.y; w[4 * q + 2] = f.z; w[4 * q + 3] = f.w;
                }
                BFrag th, tl;
#pragma unroll
                for (int q = 0; q < 16; ++q) split_bf16(w[q], th.s[q], tl.s[q]);
                bh[nt].v = th.v;
                bl[nt].v = tl.v;
            }
        }
        // 3-term split product: hi*hi + hi*lo + lo*hi  (lo*lo ~ 2^-18, dropped)
#pragma unroll
        for (int mt = 0; mt < MT; ++mt)
#pragma unroll
            for (int nt = 0; nt < NT; ++nt) {
                acc[mt][nt] = __builtin_amdgcn_wmma_f32_16x16x32_bf16(
                    false, ah[mt].v, false, bh[nt].v, (short)0, acc[mt][nt], false, false);
                acc[mt][nt] = __builtin_amdgcn_wmma_f32_16x16x32_bf16(
                    false, ah[mt].v, false, bl[nt].v, (short)0, acc[mt][nt], false, false);
                acc[mt][nt] = __builtin_amdgcn_wmma_f32_16x16x32_bf16(
                    false, al[mt].v, false, bh[nt].v, (short)0, acc[mt][nt], false, false);
            }
    }
    __syncthreads();   // everyone done reading A planes; safe to overwrite with z

    // ---- Phase 3: spill z (+bias) to LDS (C layout: VGPR r -> M = r + 8*ksel)
#pragma unroll
    for (int nt = 0; nt < NT; ++nt) {
        const int col  = ncol0 + nt * 16 + rsel;
        const float bias = fc_b[col];
#pragma unroll
        for (int mt = 0; mt < MT; ++mt)
#pragma unroll
            for (int r = 0; r < 8; ++r) {
                const int row = mt * 16 + 8 * ksel + r;
                zbuf[row * ZSTRIDE + col] = acc[mt][nt][r] + bias;
            }
    }
    __syncthreads();

    // ---- Phase 4: LN1 + sigmoid*softplus, partial mean over the 12 rows ----
    // Wave w owns rows 3w..3w+2 (all within atom w/4); writes partial slot w.
    float pacc[8];
#pragma unroll
    for (int j = 0; j < 8; ++j) pacc[j] = 0.0f;

#pragma unroll
    for (int i = 0; i < 3; ++i) {
        const int rr = wid * 3 + i;
        const float* zr = zbuf + rr * ZSTRIDE;
        float vals[16], s = 0.0f, s2 = 0.0f;
#pragma unroll
        for (int j = 0; j < 16; ++j) {
            vals[j] = zr[lane + 32 * j];
            s  += vals[j];
            s2 += vals[j] * vals[j];
        }
        s  = wave_sum(s);
        s2 = wave_sum(s2);
        const float mean = s * (1.0f / (float)KOUT);
        const float var  = s2 * (1.0f / (float)KOUT) - mean * mean;
        const float rstd = rsqrtf(var + 1e-5f);
#pragma unroll
        for (int j = 0; j < 8; ++j) {
            const int c  = lane + 32 * j;          // 0..255
            const float zg = (vals[j]     - mean) * rstd * ln1_g[c]        + ln1_b[c];
            const float zc = (vals[j + 8] - mean) * rstd * ln1_g[ADIM + c] + ln1_b[ADIM + c];
            pacc[j] += sigmoidf_(zg) * softplusf_(zc);
        }
    }
#pragma unroll
    for (int j = 0; j < 8; ++j) obuf[wid * OSTRIDE + lane + 32 * j] = pacc[j];
    __syncthreads();

    // ---- Phase 5: sum 4 partials -> mean over 12, LN2, residual add --------
    if (wid < ATOMS_PER_BLOCK) {
        const int a = wid;
        const int n = nbase + a;
        if (n < N) {
            float v[8], s = 0.0f, s2 = 0.0f;
#pragma unroll
            for (int j = 0; j < 8; ++j) {
                const int c = lane + 32 * j;
                float t = obuf[(4 * a + 0) * OSTRIDE + c] + obuf[(4 * a + 1) * OSTRIDE + c]
                        + obuf[(4 * a + 2) * OSTRIDE + c] + obuf[(4 * a + 3) * OSTRIDE + c];
                t *= (1.0f / (float)MNBR);
                v[j] = t;
                s  += t;
                s2 += t * t;
            }
            s  = wave_sum(s);
            s2 = wave_sum(s2);
            const float mean = s * (1.0f / (float)ADIM);
            const float var  = s2 * (1.0f / (float)ADIM) - mean * mean;
            const float rstd = rsqrtf(var + 1e-5f);
#pragma unroll
            for (int j = 0; j < 8; ++j) {
                const int c = lane + 32 * j;
                out[n * ADIM + c] = atom_feats[n * ADIM + c]
                                  + ((v[j] - mean) * rstd * ln2_g[c] + ln2_b[c]);
            }
        }
    }
}

extern "C" void kernel_launch(void* const* d_in, const int* in_sizes, int n_in,
                              void* d_out, int out_size, void* d_ws, size_t ws_size,
                              hipStream_t stream) {
    const float* atom_feats = (const float*)d_in[0];
    const float* bond_feats = (const float*)d_in[1];
    const float* fc_w       = (const float*)d_in[2];
    const float* fc_b       = (const float*)d_in[3];
    const float* ln1_g      = (const float*)d_in[4];
    const float* ln1_b      = (const float*)d_in[5];
    const float* ln2_g      = (const float*)d_in[6];
    const float* ln2_b      = (const float*)d_in[7];
    const int*   nbr        = (const int*)d_in[8];
    float* out = (float*)d_out;

    const int N = in_sizes[0] / ADIM;
    const int blocks = (N + ATOMS_PER_BLOCK - 1) / ATOMS_PER_BLOCK;
    const bool presplit = (ws_size >= WS_NEEDED) && (d_ws != nullptr);

    if (presplit) {
        unsigned short* w_hi = (unsigned short*)d_ws;
        unsigned short* w_lo = w_hi + WPLANE_ELEMS;

        split_fcw_kernel<<<(WPLANE_ELEMS + 255) / 256, 256, 0, stream>>>(fc_w, w_hi, w_lo);

        (void)hipFuncSetAttribute((const void*)cgcnn_conv_kernel<true>,
                                  hipFuncAttributeMaxDynamicSharedMemorySize,
                                  (int)SMEM_BYTES);
        cgcnn_conv_kernel<true><<<blocks, THREADS, SMEM_BYTES, stream>>>(
            atom_feats, bond_feats, fc_w, w_hi, w_lo, fc_b,
            ln1_g, ln1_b, ln2_g, ln2_b, nbr, out, N);
    } else {
        (void)hipFuncSetAttribute((const void*)cgcnn_conv_kernel<false>,
                                  hipFuncAttributeMaxDynamicSharedMemorySize,
                                  (int)SMEM_BYTES);
        cgcnn_conv_kernel<false><<<blocks, THREADS, SMEM_BYTES, stream>>>(
            atom_feats, bond_feats, fc_w, nullptr, nullptr, fc_b,
            ln1_g, ln1_b, ln2_g, ln2_b, nbr, out, N);
    }
}
